// GCN_LSTM_52999896432780
// MI455X (gfx1250) — matrix operand order repaired and checked by reading, not compiled, not run on hardware
//
#include <hip/hip_runtime.h>
#include <hip/hip_bf16.h>
#include <math.h>

// ---------------------------------------------------------------------------
// Problem constants (from reference)
// ---------------------------------------------------------------------------
static constexpr int kT   = 12;
static constexpr int kN   = 30000;
static constexpr int kE   = 480000;
static constexpr int kFIN = 128;
static constexpr int kH1  = 256;
static constexpr int kH2  = 128;
static constexpr int kRAW = 20;
static constexpr int kHID = 256;
static constexpr int kG4  = 4 * kHID;        // 1024
static constexpr int kD   = kH2 + kRAW;      // 148
static constexpr int kOUT = 64;

typedef __attribute__((ext_vector_type(16))) _Float16 v16h;
typedef __attribute__((ext_vector_type(8)))  _Float16 v8h;
typedef __attribute__((ext_vector_type(8)))  float    v8f;

// ---------------------------------------------------------------------------
// Generic WMMA GEMM:  C[M,Nc] = act( A[M,K] * B + addsrc + bias )
//   A f32 row-major (lda) -> f16 in LDS.
//   B f32: bT==0 -> B[k*ldb + n] (k-major), bT==1 -> B[n*ldb + k]
//   addsrc: optional side input: addsrc[row*addStride + addColOff + nGlobal]
//   act: 0 = none, 1 = relu
//
// Tile BM=64 x BN=64, BK=32; 8 waves, wave (wr,wc) owns rows wr*16..+16,
// cols wc*32..+32 => two v_wmma_f32_16x16x32_f16 per K-step.
//
// LDS layouts chosen so every fragment load is a 16-byte ds_load_b128:
//   As[r][k]  row-major   (A lane needs halves [kh*8,+8) and [16+kh*8,+8))
//   BsT[n][k] transposed  (B lane needs halves [kg*8,+8) and [16+kg*8,+8) of one column)
// Row stride 40 halves = 80 bytes (16B-aligned, bank-skewed).
// ---------------------------------------------------------------------------
#define BM 64
#define BN 64
#define BK 32
#define LDK 40

__global__ __launch_bounds__(256)
void wmma_gemm_kernel(const float* __restrict__ A,
                      const float* __restrict__ B,
                      float* __restrict__ C,
                      int M, int K, int Nc,
                      int lda, int ldb, int ldc,
                      int bT,
                      const float* __restrict__ bias,
                      const float* __restrict__ addsrc,
                      long addStride, long addColOff,
                      int act)
{
    __shared__ alignas(16) _Float16 As [BM][LDK];
    __shared__ alignas(16) _Float16 BsT[BN][LDK];

    const int tid  = threadIdx.x;
    const int lane = tid & 31;
    const int wave = tid >> 5;
    const int wr   = wave >> 1;   // 0..3 (row block of 16)
    const int wc   = wave & 1;    // 0..1 (col block of 32)

    const long row0 = (long)blockIdx.x * BM;
    const int  col0 = blockIdx.y * BN;

    v8f acc0 = {};
    v8f acc1 = {};

    // staging coordinates (fixed per thread)
    const int ar  = tid >> 2;           // A row   0..63
    const int akb = (tid & 3) * 8;      // A k-run 0,8,16,24
    const long agrow = row0 + ar;

    for (int k0 = 0; k0 < K; k0 += BK) {
        // ---- stage A tile: one 8-float run per thread ----
        {
            v8h h = {};
            if (agrow < M) {
                if (k0 + akb + 8 <= K) {
                    const float4* src =
                        reinterpret_cast<const float4*>(A + agrow * lda + k0 + akb);
                    const float4 a0 = src[0];
                    const float4 a1 = src[1];
                    h[0] = (_Float16)a0.x; h[1] = (_Float16)a0.y;
                    h[2] = (_Float16)a0.z; h[3] = (_Float16)a0.w;
                    h[4] = (_Float16)a1.x; h[5] = (_Float16)a1.y;
                    h[6] = (_Float16)a1.z; h[7] = (_Float16)a1.w;
                } else {
                    const float* src = A + agrow * lda;
#pragma unroll
                    for (int e = 0; e < 8; ++e) {
                        const int k = k0 + akb + e;
                        h[e] = (_Float16)(k < K ? src[k] : 0.0f);
                    }
                }
            }
            *reinterpret_cast<v8h*>(&As[ar][akb]) = h;
        }
        // ---- stage B tile into BsT[n][k] ----
        if (!bT) {
            // B[k,n] k-major: contiguous in n -> scalar transposed LDS stores
            const int k  = tid >> 3;            // 0..31
            const int nb = (tid & 7) * 8;       // 0..56
            if (k0 + k < K) {
                const float4* src =
                    reinterpret_cast<const float4*>(B + (long)(k0 + k) * ldb + col0 + nb);
                const float4 b0 = src[0];
                const float4 b1 = src[1];
                BsT[nb + 0][k] = (_Float16)b0.x;
                BsT[nb + 1][k] = (_Float16)b0.y;
                BsT[nb + 2][k] = (_Float16)b0.z;
                BsT[nb + 3][k] = (_Float16)b0.w;
                BsT[nb + 4][k] = (_Float16)b1.x;
                BsT[nb + 5][k] = (_Float16)b1.y;
                BsT[nb + 6][k] = (_Float16)b1.z;
                BsT[nb + 7][k] = (_Float16)b1.w;
            } else {
#pragma unroll
                for (int e = 0; e < 8; ++e) BsT[nb + e][k] = (_Float16)0.0f;
            }
        } else {
            // B[n,k]: contiguous in k -> vector path
            const int n  = tid >> 2;            // 0..63
            const int kb = (tid & 3) * 8;
            v8h h = {};
            if (k0 + kb + 8 <= K) {
                const float4* src =
                    reinterpret_cast<const float4*>(B + (long)(col0 + n) * ldb + k0 + kb);
                const float4 b0 = src[0];
                const float4 b1 = src[1];
                h[0] = (_Float16)b0.x; h[1] = (_Float16)b0.y;
                h[2] = (_Float16)b0.z; h[3] = (_Float16)b0.w;
                h[4] = (_Float16)b1.x; h[5] = (_Float16)b1.y;
                h[6] = (_Float16)b1.z; h[7] = (_Float16)b1.w;
            } else {
                const float* src = B + (long)(col0 + n) * ldb;
#pragma unroll
                for (int e = 0; e < 8; ++e) {
                    const int k = k0 + kb + e;
                    h[e] = (_Float16)(k < K ? src[k] : 0.0f);
                }
            }
            *reinterpret_cast<v8h*>(&BsT[n][kb]) = h;
        }
        __syncthreads();

        // gfx1250 global_prefetch of next A K-tile while WMMA runs
        if (k0 + BK < K && agrow < M)
            __builtin_prefetch(A + agrow * lda + k0 + BK, 0, 1);

        // ---- fragments: all 16-byte LDS loads ----
        const int khalf = lane >> 4;
        const int m     = lane & 15;
        const v8h* arow = reinterpret_cast<const v8h*>(&As[wr * 16 + m][0]);
        const v8h alo = arow[khalf];         // halves [khalf*8, +8)
        const v8h ahi = arow[2 + khalf];     // halves [16+khalf*8, +8)
        const v16h af = __builtin_shufflevector(alo, ahi,
                        0, 1, 2, 3, 4, 5, 6, 7, 8, 9, 10, 11, 12, 13, 14, 15);

        const int kg = lane >> 4;
        const int nl = lane & 15;
        const v8h* b0row = reinterpret_cast<const v8h*>(&BsT[wc * 32 + nl][0]);
        const v8h* b1row = reinterpret_cast<const v8h*>(&BsT[wc * 32 + 16 + nl][0]);
        const v16h bf0 = __builtin_shufflevector(b0row[kg], b0row[2 + kg],
                        0, 1, 2, 3, 4, 5, 6, 7, 8, 9, 10, 11, 12, 13, 14, 15);
        const v16h bf1 = __builtin_shufflevector(b1row[kg], b1row[2 + kg],
                        0, 1, 2, 3, 4, 5, 6, 7, 8, 9, 10, 11, 12, 13, 14, 15);

        acc0 = __builtin_amdgcn_wmma_f32_16x16x32_f16(false, af, false, bf0,
                                                      (short)0, acc0, false, false);
        acc1 = __builtin_amdgcn_wmma_f32_16x16x32_f16(false, af, false, bf1,
                                                      (short)0, acc1, false, false);
        __syncthreads();
    }

    // ---- epilogue: C v8f elem r -> m = r + 8*(lane>=16), n = lane&15 ----
    {
        const int half = lane >> 4;
        const int nl   = lane & 15;
#pragma unroll
        for (int r = 0; r < 8; ++r) {
            const int m = wr * 16 + r + half * 8;
            const long row = row0 + m;
            if (row >= M) continue;
            const int n0g = col0 + wc * 32 + nl;
            const int n1g = n0g + 16;
            float v0 = acc0[r];
            float v1 = acc1[r];
            if (addsrc) {
                v0 += addsrc[row * addStride + addColOff + n0g];
                v1 += addsrc[row * addStride + addColOff + n1g];
            }
            if (bias) { v0 += bias[n0g]; v1 += bias[n1g]; }
            if (act == 1) { v0 = fmaxf(v0, 0.0f); v1 = fmaxf(v1, 0.0f); }
            C[row * (long)ldc + n0g] = v0;
            C[row * (long)ldc + n1g] = v1;
        }
    }
}

// ---------------------------------------------------------------------------
// Zero fill
// ---------------------------------------------------------------------------
__global__ void fill_zero_kernel(float* __restrict__ p, long n)
{
    long i = (long)blockIdx.x * blockDim.x + threadIdx.x;
    if (i < n) p[i] = 0.0f;
}

// ---------------------------------------------------------------------------
// spmm scatter-add: out[t, row[e], :] += w[e] * sup[t, col[e], :]
// One block per edge (grid.y = t), one lane per feature (coalesced).
// ---------------------------------------------------------------------------
__global__ void spmm_kernel(const float* __restrict__ sup,
                            const int* __restrict__ rows,
                            const int* __restrict__ cols,
                            const float* __restrict__ w,
                            float* __restrict__ out,
                            int H, int E, int Nn)
{
    const long e = blockIdx.x;
    const int  t = blockIdx.y;
    const long ei = (long)t * E + e;
    const int r = rows[ei];
    const int c = cols[ei];
    const float ww = w[ei];
    const int f = threadIdx.x;
    atomicAdd(&out[((long)t * Nn + r) * H + f],
              ww * sup[((long)t * Nn + c) * H + f]);
}

// ---------------------------------------------------------------------------
// data[i] = act(data[i] + bias[i % H])
// ---------------------------------------------------------------------------
__global__ void bias_act_kernel(float* __restrict__ data,
                                const float* __restrict__ bias,
                                long total, int H, int act)
{
    long i = (long)blockIdx.x * blockDim.x + threadIdx.x;
    if (i >= total) return;
    float v = data[i] + bias[i % H];
    if (act) v = fmaxf(v, 0.0f);
    data[i] = v;
}

// ---------------------------------------------------------------------------
// seq[n, t, :] = concat(embed[t, n, :kH2], raw[t, n, :kRAW])
// ---------------------------------------------------------------------------
__global__ void seq_concat_kernel(const float* __restrict__ embed,
                                  const float* __restrict__ raw,
                                  float* __restrict__ seq)
{
    long i = (long)blockIdx.x * blockDim.x + threadIdx.x;
    const long total = (long)kN * kT * kD;
    if (i >= total) return;
    const long n   = i / ((long)kT * kD);
    const int  rem = (int)(i - n * (long)kT * kD);
    const int  t   = rem / kD;
    const int  k   = rem % kD;
    float v;
    if (k < kH2) v = embed[((long)t * kN + n) * kH2 + k];
    else         v = raw[((long)t * kN + n) * kRAW + (k - kH2)];
    seq[i] = v;
}

// ---------------------------------------------------------------------------
// LSTM gates (torch order i,f,g,o): updates h,c and stores h into hs[:,t,:]
// ---------------------------------------------------------------------------
__device__ __forceinline__ float sigmoidf(float x) { return 1.0f / (1.0f + __expf(-x)); }

__global__ void lstm_gates_kernel(const float* __restrict__ g,
                                  float* __restrict__ h,
                                  float* __restrict__ c,
                                  float* __restrict__ hs,
                                  int t)
{
    long idx = (long)blockIdx.x * blockDim.x + threadIdx.x;
    if (idx >= (long)kN * kHID) return;
    const long n = idx / kHID;
    const int  j = (int)(idx % kHID);
    const float* gr = g + n * (long)kG4;
    const float gi = sigmoidf(gr[j]);
    const float gf = sigmoidf(gr[kHID + j]);
    const float gg = tanhf(gr[2 * kHID + j]);
    const float go = sigmoidf(gr[3 * kHID + j]);
    const float cn = gf * c[idx] + gi * gg;
    const float hn = go * tanhf(cn);
    c[idx] = cn;
    h[idx] = hn;
    hs[(n * kT + t) * (long)kHID + j] = hn;
}

// ---------------------------------------------------------------------------
// Attention pooling over time: one block per node.
// ---------------------------------------------------------------------------
__global__ __launch_bounds__(256)
void attention_kernel(const float* __restrict__ hs,
                      const float* __restrict__ att_w,
                      const float* __restrict__ att_b,
                      float* __restrict__ pooled)
{
    const int n = blockIdx.x;
    __shared__ float sc[16];
    const float* hn = hs + (long)n * kT * kHID;
    const int tid = threadIdx.x;

    if (tid < kT) {
        float s = att_b[0];
        for (int j = 0; j < kHID; ++j)
            s += hn[tid * kHID + j] * att_w[j];
        sc[tid] = s;
    }
    __syncthreads();
    if (tid == 0) {
        float m = sc[0];
        for (int t = 1; t < kT; ++t) m = fmaxf(m, sc[t]);
        float ssum = 0.0f;
        for (int t = 0; t < kT; ++t) { float e = __expf(sc[t] - m); sc[t] = e; ssum += e; }
        const float inv = 1.0f / ssum;
        for (int t = 0; t < kT; ++t) sc[t] *= inv;
    }
    __syncthreads();
    for (int j = tid; j < kHID; j += blockDim.x) {
        float acc = 0.0f;
        for (int t = 0; t < kT; ++t) acc += sc[t] * hn[t * kHID + j];
        pooled[(long)n * kHID + j] = acc;
    }
}

// ---------------------------------------------------------------------------
// Host-side orchestration
// ---------------------------------------------------------------------------
static inline int cdiv(long a, long b) { return (int)((a + b - 1) / b); }

extern "C" void kernel_launch(void* const* d_in, const int* in_sizes, int n_in,
                              void* d_out, int out_size, void* d_ws, size_t ws_size,
                              hipStream_t stream)
{
    (void)in_sizes; (void)n_in; (void)out_size; (void)ws_size;

    const float* nodes    = (const float*)d_in[0];   // [T,N,128]
    const int*   edge_row = (const int*)  d_in[1];   // [T,E]
    const int*   edge_col = (const int*)  d_in[2];   // [T,E]
    const float* edge_w   = (const float*)d_in[3];   // [T,E]
    const float* raw      = (const float*)d_in[4];   // [T,N,20]
    const float* W1   = (const float*)d_in[5];
    const float* b1   = (const float*)d_in[6];
    const float* W2   = (const float*)d_in[7];
    const float* b2   = (const float*)d_in[8];
    const float* Wih0 = (const float*)d_in[9];       // [1024,148]
    const float* Whh0 = (const float*)d_in[10];      // [1024,256]
    const float* bih0 = (const float*)d_in[11];
    const float* bhh0 = (const float*)d_in[12];
    const float* Wih1 = (const float*)d_in[13];      // [1024,256]
    const float* Whh1 = (const float*)d_in[14];      // [1024,256]
    const float* bih1 = (const float*)d_in[15];
    const float* bhh1 = (const float*)d_in[16];
    const float* att_w = (const float*)d_in[17];
    const float* att_b = (const float*)d_in[18];
    const float* fcn_w = (const float*)d_in[19];     // [256,128]
    const float* fcn_b = (const float*)d_in[20];
    const float* out_w = (const float*)d_in[21];     // [128,64]
    const float* out_b = (const float*)d_in[22];
    float* out = (float*)d_out;                      // [N,64]

    // ---- workspace carve-up (floats) ----
    float* ws = (float*)d_ws;
    size_t off = 0;
    auto carve = [&](size_t nelem) { float* p = ws + off; off += nelem; return p; };

    const long TN = (long)kT * kN;       // 360000 rows
    float* sup1  = carve(TN * kH1);      // GEMM1 out        [T*N,256]
    float* xacc  = carve(TN * kH1);      // spmm1 acc / x    [T*N,256]
    float* sup2  = carve(TN * kH2);      // GEMM2 out        [T*N,128]
    float* embed = carve(TN * kH2);      // spmm2 acc / embed[T*N,128]
    float* seq   = carve((long)kN * kT * kD);   // [N,T,148]
    float* xg    = carve((long)kN * kT * kG4);  // [N,T,1024] (reused layer 1)
    float* hs0   = carve((long)kN * kT * kHID); // [N,T,256]
    float* hs1   = carve((long)kN * kT * kHID); // [N,T,256]
    float* gbuf  = carve((long)kN * kG4);       // [N,1024]
    float* hbuf  = carve((long)kN * kHID);
    float* cbuf  = carve((long)kN * kHID);
    float* pooled= carve((long)kN * kHID);
    float* zbuf  = carve((long)kN * 128);

    const dim3 blk256(256);

    // ================= GCN =================
    // sup1 = nodes @ W1
    wmma_gemm_kernel<<<dim3(cdiv(TN, BM), kH1 / BN), blk256, 0, stream>>>(
        nodes, W1, sup1, (int)TN, kFIN, kH1, kFIN, kH1, kH1,
        /*bT*/0, nullptr, nullptr, 0, 0, /*act*/0);

    // xacc = spmm(sup1); x = relu(xacc + b1)
    fill_zero_kernel<<<cdiv(TN * kH1, 256), blk256, 0, stream>>>(xacc, TN * kH1);
    spmm_kernel<<<dim3(kE, kT), dim3(kH1), 0, stream>>>(
        sup1, edge_row, edge_col, edge_w, xacc, kH1, kE, kN);
    bias_act_kernel<<<cdiv(TN * kH1, 256), blk256, 0, stream>>>(
        xacc, b1, TN * kH1, kH1, /*relu*/1);

    // sup2 = x @ W2
    wmma_gemm_kernel<<<dim3(cdiv(TN, BM), kH2 / BN), blk256, 0, stream>>>(
        xacc, W2, sup2, (int)TN, kH1, kH2, kH1, kH2, kH2,
        0, nullptr, nullptr, 0, 0, 0);

    // embed = spmm(sup2) + b2
    fill_zero_kernel<<<cdiv(TN * kH2, 256), blk256, 0, stream>>>(embed, TN * kH2);
    spmm_kernel<<<dim3(kE, kT), dim3(kH2), 0, stream>>>(
        sup2, edge_row, edge_col, edge_w, embed, kH2, kE, kN);
    bias_act_kernel<<<cdiv(TN * kH2, 256), blk256, 0, stream>>>(
        embed, b2, TN * kH2, kH2, 0);

    // seq[N,T,148] = concat(embed, raw) transposed to batch-first
    seq_concat_kernel<<<cdiv((long)kN * kT * kD, 256), blk256, 0, stream>>>(embed, raw, seq);

    // ================= LSTM layer 0 =================
    // xg = seq @ Wih0^T + bih0   [N*T, 1024]
    wmma_gemm_kernel<<<dim3(cdiv(TN, BM), kG4 / BN), blk256, 0, stream>>>(
        seq, Wih0, xg, (int)TN, kD, kG4, kD, kD, kG4,
        /*bT*/1, bih0, nullptr, 0, 0, 0);

    fill_zero_kernel<<<cdiv((long)kN * kHID, 256), blk256, 0, stream>>>(hbuf, (long)kN * kHID);
    fill_zero_kernel<<<cdiv((long)kN * kHID, 256), blk256, 0, stream>>>(cbuf, (long)kN * kHID);
    for (int t = 0; t < kT; ++t) {
        // g = xg[:,t,:] + h @ Whh0^T + bhh0
        wmma_gemm_kernel<<<dim3(cdiv(kN, BM), kG4 / BN), blk256, 0, stream>>>(
            hbuf, Whh0, gbuf, kN, kHID, kG4, kHID, kHID, kG4,
            /*bT*/1, bhh0, xg, (long)kT * kG4, (long)t * kG4, 0);
        lstm_gates_kernel<<<cdiv((long)kN * kHID, 256), blk256, 0, stream>>>(
            gbuf, hbuf, cbuf, hs0, t);
    }

    // ================= LSTM layer 1 =================
    // xg = hs0 @ Wih1^T + bih1   (reuse xg buffer)
    wmma_gemm_kernel<<<dim3(cdiv(TN, BM), kG4 / BN), blk256, 0, stream>>>(
        hs0, Wih1, xg, (int)TN, kHID, kG4, kHID, kHID, kG4,
        1, bih1, nullptr, 0, 0, 0);

    fill_zero_kernel<<<cdiv((long)kN * kHID, 256), blk256, 0, stream>>>(hbuf, (long)kN * kHID);
    fill_zero_kernel<<<cdiv((long)kN * kHID, 256), blk256, 0, stream>>>(cbuf, (long)kN * kHID);
    for (int t = 0; t < kT; ++t) {
        wmma_gemm_kernel<<<dim3(cdiv(kN, BM), kG4 / BN), blk256, 0, stream>>>(
            hbuf, Whh1, gbuf, kN, kHID, kG4, kHID, kHID, kG4,
            1, bhh1, xg, (long)kT * kG4, (long)t * kG4, 0);
        lstm_gates_kernel<<<cdiv((long)kN * kHID, 256), blk256, 0, stream>>>(
            gbuf, hbuf, cbuf, hs1, t);
    }

    // ================= attention pooling =================
    attention_kernel<<<dim3(kN), blk256, 0, stream>>>(hs1, att_w, att_b, pooled);

    // ================= FCN head =================
    // z = relu(pooled @ fcn_w + fcn_b)
    wmma_gemm_kernel<<<dim3(cdiv(kN, BM), 128 / BN), blk256, 0, stream>>>(
        pooled, fcn_w, zbuf, kN, kHID, 128, kHID, 128, 128,
        0, fcn_b, nullptr, 0, 0, /*relu*/1);
    // out = z @ out_w + out_b
    wmma_gemm_kernel<<<dim3(cdiv(kN, BM), kOUT / BN), blk256, 0, stream>>>(
        zbuf, out_w, out, kN, 128, kOUT, 128, kOUT, kOUT,
        0, out_b, nullptr, 0, 0, 0);
}
